// TestNet_24257975287983
// MI455X (gfx1250) — matrix operand order repaired and checked by reading, compiled, not run on hardware
//
#include <hip/hip_runtime.h>
#include <stdint.h>

// ---------------- problem constants ----------------
#define N_NODES 100000
#define N_EDGES 1200000
#define F1 64
#define F2 128
#define F3 256
#define KP1 4096
#define KP2 256
#define M1PAD 131072   // next pow2 >= N_NODES for bitonic sort
#define NEG_SLOPE 0.01f

typedef __attribute__((ext_vector_type(16))) __bf16 v16bf;
typedef __attribute__((ext_vector_type(8)))  float  v8f;

// ---------------- workspace layout (bytes, all 128-aligned) ----------------
static constexpr size_t A1_OFF   = 0;                                  // x1 / h1 [N,64]
static constexpr size_t B1_OFF   = A1_OFF   + (size_t)N_NODES*F1*4;    // agg1    [N,64]
static constexpr size_t DEG1_OFF = B1_OFF   + (size_t)N_NODES*F1*4;    // deg     [N]
static constexpr size_t KEYS_OFF = DEG1_OFF + (size_t)N_NODES*4;       // u64 keys[131072]
static constexpr size_t MAP_OFF  = KEYS_OFF + (size_t)M1PAD*8;         // int [N]
static constexpr size_t KEPT_OFF = MAP_OFF  + (size_t)N_NODES*4;       // int [N]
static constexpr size_t SRC2_OFF = KEPT_OFF + (size_t)N_NODES*4;       // int [E]
static constexpr size_t DST2_OFF = SRC2_OFF + (size_t)N_EDGES*4;       // int [E]
static constexpr size_t WE_OFF   = DST2_OFF + (size_t)N_EDGES*4;       // f32 [E]
static constexpr size_t HP1_OFF  = WE_OFF   + (size_t)N_EDGES*4;       // [4096,64]
static constexpr size_t X2_OFF   = HP1_OFF  + (size_t)KP1*F1*4;        // [4096,128]
static constexpr size_t AGG2_OFF = X2_OFF   + (size_t)KP1*F2*4;        // [4096,128] -> h2
static constexpr size_t DEG2_OFF = AGG2_OFF + (size_t)KP1*F2*4;        // [4096]
static constexpr size_t X3_OFF   = DEG2_OFF + (size_t)KP1*4;           // [4096,256]
static constexpr size_t AGG3_OFF = X3_OFF   + (size_t)KP1*F3*4;        // [4096,256] -> h3
static constexpr size_t MAP2_OFF = AGG3_OFF + (size_t)KP1*F3*4;        // int [4096]
static constexpr size_t KEPT2_OFF= MAP2_OFF + (size_t)KP1*4;           // int [4096]
static constexpr size_t X4_OFF   = KEPT2_OFF+ (size_t)KP1*4;           // [256,256]
static constexpr size_t PN_OFF   = X4_OFF   + (size_t)KP2*F3*4;        // 2 scalars

// ---------------- small utility kernels ----------------
__global__ void fill_u32_k(unsigned* p, unsigned v, long long n) {
  long long stride = (long long)gridDim.x * blockDim.x;
  for (long long i = (long long)blockIdx.x * blockDim.x + threadIdx.x; i < n; i += stride)
    p[i] = v;
}

// x1 = pos @ W1   ([N,3] x [3,64])
__global__ void gcn1_xw_k(const float* __restrict__ pos, const float* __restrict__ W1,
                          float* __restrict__ x1, long long total) {
  long long stride = (long long)gridDim.x * blockDim.x;
  for (long long i = (long long)blockIdx.x * blockDim.x + threadIdx.x; i < total; i += stride) {
    int node = (int)(i >> 6), c = (int)(i & 63);
    float s = pos[node * 3 + 0] * W1[c] + pos[node * 3 + 1] * W1[64 + c] +
              pos[node * 3 + 2] * W1[128 + c];
    x1[i] = s;
  }
}

// deg[dst] += 1 over int64 edge list
__global__ void deg1_k(const long long* __restrict__ ei, float* __restrict__ deg, int E) {
  long long stride = (long long)gridDim.x * blockDim.x;
  for (long long e = (long long)blockIdx.x * blockDim.x + threadIdx.x; e < E; e += stride)
    atomicAdd(&deg[(int)ei[E + e]], 1.0f);
}

// agg[dst, 4c..4c+3] += x[src, 4c..4c+3] * dinv[src]*dinv[dst]  (w == 1, float4 gather)
__global__ void scatter1_k(const long long* __restrict__ ei, const float* __restrict__ deg,
                           const float* __restrict__ x, float* __restrict__ agg,
                           long long total, int E) {
  long long stride = (long long)gridDim.x * blockDim.x;
  for (long long i = (long long)blockIdx.x * blockDim.x + threadIdx.x; i < total; i += stride) {
    int e = (int)(i >> 4), cg = (int)(i & 15);     // 16 groups of 4 -> F1=64
    int s = (int)ei[e], d = (int)ei[E + e];
    float nrm = rsqrtf(deg[s] + 1.0f) * rsqrtf(deg[d] + 1.0f);
    float4 xv = *(const float4*)&x[((long long)s << 6) + (cg << 2)];
    float* ap = &agg[((long long)d << 6) + (cg << 2)];
    atomicAdd(ap + 0, xv.x * nrm);
    atomicAdd(ap + 1, xv.y * nrm);
    atomicAdd(ap + 2, xv.z * nrm);
    atomicAdd(ap + 3, xv.w * nrm);
  }
}

// out = leaky(agg + x/(deg+1) + b)   (out may alias x or agg elementwise)
__global__ void combine_k(const float* __restrict__ x, const float* __restrict__ agg,
                          const float* __restrict__ deg, const float* __restrict__ bias,
                          float* __restrict__ out, long long total, int fshift) {
  long long stride = (long long)gridDim.x * blockDim.x;
  int fmask = (1 << fshift) - 1;
  for (long long i = (long long)blockIdx.x * blockDim.x + threadIdx.x; i < total; i += stride) {
    int node = (int)(i >> fshift), c = (int)(i & fmask);
    float v = agg[i] + x[i] / (deg[node] + 1.0f) + bias[c];
    out[i] = v >= 0.0f ? v : NEG_SLOPE * v;
  }
}

// ||p||  (single block)
__global__ void pnorm_k(const float* __restrict__ p, int n, float* __restrict__ out) {
  __shared__ float red[256];
  float s = 0.0f;
  for (int i = threadIdx.x; i < n; i += 256) s += p[i] * p[i];
  red[threadIdx.x] = s;
  __syncthreads();
  for (int off = 128; off > 0; off >>= 1) {
    if ((int)threadIdx.x < off) red[threadIdx.x] += red[threadIdx.x + off];
    __syncthreads();
  }
  if (threadIdx.x == 0) out[0] = sqrtf(red[0]);
}

// keys[n] = (orderkey(tanh(h.p/||p||)) << 32) | ~n   -> sort descending == top-k w/ JAX tiebreak
__global__ void score_k(const float* __restrict__ h, const float* __restrict__ p,
                        const float* __restrict__ pnorm, unsigned long long* __restrict__ keys,
                        int n, int F) {
  long long stride = (long long)gridDim.x * blockDim.x;
  for (long long node = (long long)blockIdx.x * blockDim.x + threadIdx.x; node < n; node += stride) {
    float s = 0.0f;
    for (int c = 0; c < F; ++c) s += h[node * F + c] * p[c];
    float sc = tanhf(s / pnorm[0]);
    unsigned u = __float_as_uint(sc);
    u = (u & 0x80000000u) ? ~u : (u | 0x80000000u);
    keys[node] = ((unsigned long long)u << 32) | (unsigned)(0xFFFFFFFFu - (unsigned)node);
  }
}

// one step of a descending bitonic sort
__global__ void bitonic_k(unsigned long long* __restrict__ keys, int j, int k, int M) {
  int i = blockIdx.x * blockDim.x + threadIdx.x;
  if (i >= M) return;
  int ixj = i ^ j;
  if (ixj > i) {
    unsigned long long a = keys[i], b = keys[ixj];
    bool up = ((i & k) == 0);
    if ((up && a < b) || (!up && a > b)) { keys[i] = b; keys[ixj] = a; }
  }
}

__global__ void mark_k(const unsigned long long* __restrict__ keys, int* __restrict__ mapping,
                       int* __restrict__ kept, int k) {
  int i = blockIdx.x * blockDim.x + threadIdx.x;
  if (i < k) {
    int idx = (int)(0xFFFFFFFFu - (unsigned)keys[i]);
    mapping[idx] = i;
    kept[idx] = 1;
  }
}

// x_new[i,c] = h[perm[i],c] * score_sorted[i]
__global__ void gather_k(const unsigned long long* __restrict__ keys, const float* __restrict__ h,
                         float* __restrict__ xo, long long total, int fshift) {
  long long stride = (long long)gridDim.x * blockDim.x;
  int fmask = (1 << fshift) - 1;
  for (long long i = (long long)blockIdx.x * blockDim.x + threadIdx.x; i < total; i += stride) {
    int r = (int)(i >> fshift), c = (int)(i & fmask);
    unsigned long long key = keys[r];
    unsigned u = (unsigned)(key >> 32);
    unsigned bits = (u & 0x80000000u) ? (u ^ 0x80000000u) : ~u;
    float sc = __uint_as_float(bits);
    int idx = (int)(0xFFFFFFFFu - (unsigned)key);
    xo[i] = h[(long long)idx << fshift | c] * sc;
  }
}

__global__ void remap_k(const long long* __restrict__ ei, const int* __restrict__ mapping,
                        const int* __restrict__ kept, int* __restrict__ src2,
                        int* __restrict__ dst2, float* __restrict__ we, int E) {
  long long stride = (long long)gridDim.x * blockDim.x;
  for (long long e = (long long)blockIdx.x * blockDim.x + threadIdx.x; e < E; e += stride) {
    int s = (int)ei[e], d = (int)ei[E + e];
    src2[e] = mapping[s];
    dst2[e] = mapping[d];
    we[e] = (kept[s] && kept[d]) ? 1.0f : 0.0f;
  }
}

__global__ void deg_k(const int* __restrict__ dst, const float* __restrict__ we,
                      float* __restrict__ deg, int E) {
  long long stride = (long long)gridDim.x * blockDim.x;
  for (long long e = (long long)blockIdx.x * blockDim.x + threadIdx.x; e < E; e += stride)
    atomicAdd(&deg[dst[e]], we[e]);
}

// vectorized scatter: one thread per (edge, 4-feature group)
__global__ void scatter_k(const int* __restrict__ src, const int* __restrict__ dst,
                          const float* __restrict__ we, const float* __restrict__ deg,
                          const float* __restrict__ x, float* __restrict__ agg,
                          long long total, int fshift) {
  long long stride = (long long)gridDim.x * blockDim.x;
  int gshift = fshift - 2;
  int gmask = (1 << gshift) - 1;
  for (long long i = (long long)blockIdx.x * blockDim.x + threadIdx.x; i < total; i += stride) {
    int e = (int)(i >> gshift), cg = (int)(i & gmask);
    int s = src[e], d = dst[e];
    float nrm = rsqrtf(deg[s] + 1.0f) * we[e] * rsqrtf(deg[d] + 1.0f);
    float4 xv = *(const float4*)&x[((long long)s << fshift) + (cg << 2)];
    float* ap = &agg[((long long)d << fshift) + (cg << 2)];
    atomicAdd(ap + 0, xv.x * nrm);
    atomicAdd(ap + 1, xv.y * nrm);
    atomicAdd(ap + 2, xv.z * nrm);
    atomicAdd(ap + 3, xv.w * nrm);
  }
}

// ---------------- WMMA GEMM: C[M,N] = A[M,K] * B[K,N] ----------------
// f32 in memory, bf16 matrix cores, f32 accum. LDS-staged tiles, fully unrolled K.
// Block: 256 threads (8 waves) computes a 64x32 C tile; wave grid 4x2 of 16x16 WMMA tiles.
__device__ __forceinline__ unsigned pkbf(float lo, float hi) {
  union { __bf16 b[2]; unsigned u; } t;
  t.b[0] = (__bf16)lo;
  t.b[1] = (__bf16)hi;
  return t.u;
}

template <int K, int N>
__global__ void wmma_gemm_bf16_k(const float* __restrict__ A, const float* __restrict__ B,
                                 float* __restrict__ C) {
  __shared__ __bf16 Abuf[64 * 32];   // [row][k]  (64 rows x 32 k)
  __shared__ __bf16 Bbuf[32 * 32];   // [n][k]    (transposed: 32 n x 32 k)

  const int t = threadIdx.x;
  const int nbn = N / 32;
  const int m0 = (blockIdx.x / nbn) * 64;
  const int n0 = (blockIdx.x % nbn) * 32;

  const int wid = t >> 5;
  const int wave_m = wid >> 1;             // 0..3
  const int wave_n = wid & 1;              // 0..1
  const int lane = t & 31;
  const int g = lane >> 4;                 // half-wave
  const int l = lane & 15;

  // staging coordinates
  const int arow = t >> 2, aseg = t & 3;   // A: 64 rows x 4 segs of 8 floats
  const int bk = t >> 3, bnseg = t & 7;    // B: 32 k x 8 segs of 4 n

  v8f acc = {};
#pragma unroll
  for (int k0 = 0; k0 < K; k0 += 32) {
    __syncthreads();  // previous iteration's LDS reads complete
    // ---- stage A tile (64x32 f32 -> bf16), coalesced b128 loads ----
    {
      const float* Ag = A + (size_t)(m0 + arow) * K + k0 + aseg * 8;
      float4 a0 = *(const float4*)(Ag);
      float4 a1 = *(const float4*)(Ag + 4);
      uint4 pk;
      pk.x = pkbf(a0.x, a0.y); pk.y = pkbf(a0.z, a0.w);
      pk.z = pkbf(a1.x, a1.y); pk.w = pkbf(a1.z, a1.w);
      *(uint4*)&Abuf[arow * 32 + aseg * 8] = pk;
    }
    // ---- stage B tile (32x32) transposed into [n][k] ----
    {
      const float* Bg = B + (size_t)(k0 + bk) * N + n0 + bnseg * 4;
      float4 b4 = *(const float4*)Bg;
      __bf16* Bp = &Bbuf[(bnseg * 4) * 32 + bk];
      Bp[0]  = (__bf16)b4.x;
      Bp[32] = (__bf16)b4.y;
      Bp[64] = (__bf16)b4.z;
      Bp[96] = (__bf16)b4.w;
    }
    __syncthreads();

    // ---- per-lane fragments: contiguous 16B LDS reads ----
    // A (16x32): lanes 0-15 M=l, K={g*8..g*8+7, 16+g*8..16+g*8+7}
    union Frag { uint4 q[2]; v16bf v; };
    Frag af, bf;
    const __bf16* Ar = &Abuf[(wave_m * 16 + l) * 32 + (g << 3)];
    af.q[0] = *(const uint4*)(Ar);
    af.q[1] = *(const uint4*)(Ar + 16);
    // B (32x16): lane's N = l, K = {g*8..g*8+7, 16+g*8..16+g*8+7} (matches A)
    const __bf16* Br = &Bbuf[(wave_n * 16 + l) * 32 + (g << 3)];
    bf.q[0] = *(const uint4*)(Br);
    bf.q[1] = *(const uint4*)(Br + 16);

    acc = __builtin_amdgcn_wmma_f32_16x16x32_bf16(false, af.v, false, bf.v, (short)0, acc,
                                                  false, false);
  }

#pragma unroll
  for (int r = 0; r < 8; ++r)  // C/D: VGPR r -> M = r + 8*g, N = l
    C[(size_t)(m0 + wave_m * 16 + r + (g << 3)) * N + (n0 + wave_n * 16 + l)] = acc[r];
}

// ---------------- final FC: out[512] = fcb + vec(x4^T) @ fcW ----------------
__global__ void init_out_k(const float* __restrict__ fcb, float* __restrict__ out) {
  out[threadIdx.x] = fcb[threadIdx.x];
}

// split-K GEMV: each block handles 512 K-rows; 128 threads x float4 columns
__global__ void fc_k(const float* __restrict__ x4, const float* __restrict__ fcW,
                     float* __restrict__ out) {
  __shared__ float vs[512];
  int t = threadIdx.x;                       // 0..127
  int k0 = blockIdx.x * 512;
  for (int r = t; r < 512; r += 128) {
    int k = k0 + r;                          // flatten(x.T): k = feat*256 + node
    vs[r] = x4[((k & 255) << 8) | (k >> 8)];
  }
  __syncthreads();
  float4 acc = {0.0f, 0.0f, 0.0f, 0.0f};
  const float* Wp = fcW + (long long)k0 * 512 + (t << 2);
  for (int r = 0; r < 512; ++r) {
    __builtin_prefetch(Wp + (long long)(r + 16) * 512, 0, 0);
    float4 w = *(const float4*)(Wp + (long long)r * 512);
    float v = vs[r];
    acc.x += v * w.x; acc.y += v * w.y; acc.z += v * w.z; acc.w += v * w.w;
  }
  float* op = &out[t << 2];
  atomicAdd(op + 0, acc.x);
  atomicAdd(op + 1, acc.y);
  atomicAdd(op + 2, acc.z);
  atomicAdd(op + 3, acc.w);
}

// ---------------- host orchestration ----------------
static inline int nb(long long n, int bs) {
  long long b = (n + bs - 1) / bs;
  return (int)(b > 16384 ? 16384 : b);
}

extern "C" void kernel_launch(void* const* d_in, const int* in_sizes, int n_in,
                              void* d_out, int out_size, void* d_ws, size_t ws_size,
                              hipStream_t stream) {
  const float*     pos = (const float*)d_in[0];
  const long long* ei  = (const long long*)d_in[1];
  const float* W1  = (const float*)d_in[2];
  const float* b1  = (const float*)d_in[3];
  const float* W2  = (const float*)d_in[4];
  const float* b2  = (const float*)d_in[5];
  const float* W3  = (const float*)d_in[6];
  const float* b3  = (const float*)d_in[7];
  const float* p1  = (const float*)d_in[8];
  const float* p2  = (const float*)d_in[9];
  const float* fcW = (const float*)d_in[10];
  const float* fcb = (const float*)d_in[11];
  float* out = (float*)d_out;
  char*  ws  = (char*)d_ws;

  float* A1   = (float*)(ws + A1_OFF);
  float* B1   = (float*)(ws + B1_OFF);
  float* DEG1 = (float*)(ws + DEG1_OFF);
  unsigned long long* KEYS = (unsigned long long*)(ws + KEYS_OFF);
  int*   MAP  = (int*)(ws + MAP_OFF);
  int*   KEPT = (int*)(ws + KEPT_OFF);
  int*   SRC2 = (int*)(ws + SRC2_OFF);
  int*   DST2 = (int*)(ws + DST2_OFF);
  float* WE   = (float*)(ws + WE_OFF);
  float* HP1  = (float*)(ws + HP1_OFF);
  float* X2   = (float*)(ws + X2_OFF);
  float* AGG2 = (float*)(ws + AGG2_OFF);
  float* DEG2 = (float*)(ws + DEG2_OFF);
  float* X3   = (float*)(ws + X3_OFF);
  float* AGG3 = (float*)(ws + AGG3_OFF);
  int*   MAP2 = (int*)(ws + MAP2_OFF);
  int*   KEPT2= (int*)(ws + KEPT2_OFF);
  float* X4   = (float*)(ws + X4_OFF);
  float* PN   = (float*)(ws + PN_OFF);

  // ---- zero all accumulators / flags (must happen every call) ----
  fill_u32_k<<<nb((long long)N_NODES * F1, 256), 256, 0, stream>>>((unsigned*)B1, 0, (long long)N_NODES * F1);
  fill_u32_k<<<nb(N_NODES, 256), 256, 0, stream>>>((unsigned*)DEG1, 0, N_NODES);
  fill_u32_k<<<nb((long long)M1PAD * 2, 256), 256, 0, stream>>>((unsigned*)KEYS, 0, (long long)M1PAD * 2);
  fill_u32_k<<<nb(N_NODES, 256), 256, 0, stream>>>((unsigned*)MAP, 0, N_NODES);
  fill_u32_k<<<nb(N_NODES, 256), 256, 0, stream>>>((unsigned*)KEPT, 0, N_NODES);
  fill_u32_k<<<nb((long long)KP1 * F2, 256), 256, 0, stream>>>((unsigned*)AGG2, 0, (long long)KP1 * F2);
  fill_u32_k<<<nb((long long)KP1 * F3, 256), 256, 0, stream>>>((unsigned*)AGG3, 0, (long long)KP1 * F3);
  fill_u32_k<<<nb(KP1, 256), 256, 0, stream>>>((unsigned*)DEG2, 0, KP1);
  fill_u32_k<<<nb(KP1, 256), 256, 0, stream>>>((unsigned*)MAP2, 0, KP1);
  fill_u32_k<<<nb(KP1, 256), 256, 0, stream>>>((unsigned*)KEPT2, 0, KP1);

  // ---- GCN1 ----
  gcn1_xw_k<<<nb((long long)N_NODES * F1, 256), 256, 0, stream>>>(pos, W1, A1, (long long)N_NODES * F1);
  deg1_k<<<nb(N_EDGES, 256), 256, 0, stream>>>(ei, DEG1, N_EDGES);
  scatter1_k<<<nb((long long)N_EDGES * 16, 256), 256, 0, stream>>>(ei, DEG1, A1, B1,
                                                                   (long long)N_EDGES * 16, N_EDGES);
  combine_k<<<nb((long long)N_NODES * F1, 256), 256, 0, stream>>>(A1, B1, DEG1, b1, A1,
                                                                  (long long)N_NODES * F1, 6);

  // ---- TopKPool 1 (k=4096) ----
  pnorm_k<<<1, 256, 0, stream>>>(p1, F1, PN);
  score_k<<<nb(N_NODES, 256), 256, 0, stream>>>(A1, p1, PN, KEYS, N_NODES, F1);
  for (int k = 2; k <= M1PAD; k <<= 1)
    for (int j = k >> 1; j > 0; j >>= 1)
      bitonic_k<<<M1PAD / 256, 256, 0, stream>>>(KEYS, j, k, M1PAD);
  mark_k<<<KP1 / 256, 256, 0, stream>>>(KEYS, MAP, KEPT, KP1);
  gather_k<<<nb((long long)KP1 * F1, 256), 256, 0, stream>>>(KEYS, A1, HP1, (long long)KP1 * F1, 6);
  remap_k<<<nb(N_EDGES, 256), 256, 0, stream>>>(ei, MAP, KEPT, SRC2, DST2, WE, N_EDGES);

  // ---- GCN2 (WMMA 4096x64x128) ----
  deg_k<<<nb(N_EDGES, 256), 256, 0, stream>>>(DST2, WE, DEG2, N_EDGES);
  wmma_gemm_bf16_k<F1, F2><<<(KP1 / 64) * (F2 / 32), 256, 0, stream>>>(HP1, W2, X2);
  scatter_k<<<nb((long long)N_EDGES * (F2 / 4), 256), 256, 0, stream>>>(
      SRC2, DST2, WE, DEG2, X2, AGG2, (long long)N_EDGES * (F2 / 4), 7);
  combine_k<<<nb((long long)KP1 * F2, 256), 256, 0, stream>>>(X2, AGG2, DEG2, b2, AGG2,
                                                              (long long)KP1 * F2, 7);

  // ---- GCN3 (WMMA 4096x128x256) ----
  wmma_gemm_bf16_k<F2, F3><<<(KP1 / 64) * (F3 / 32), 256, 0, stream>>>(AGG2, W3, X3);
  scatter_k<<<nb((long long)N_EDGES * (F3 / 4), 256), 256, 0, stream>>>(
      SRC2, DST2, WE, DEG2, X3, AGG3, (long long)N_EDGES * (F3 / 4), 8);
  combine_k<<<nb((long long)KP1 * F3, 256), 256, 0, stream>>>(X3, AGG3, DEG2, b3, AGG3,
                                                              (long long)KP1 * F3, 8);

  // ---- TopKPool 2 (k=256) ----
  pnorm_k<<<1, 256, 0, stream>>>(p2, F3, PN + 1);
  score_k<<<nb(KP1, 256), 256, 0, stream>>>(AGG3, p2, PN + 1, KEYS, KP1, F3);
  for (int k = 2; k <= KP1; k <<= 1)
    for (int j = k >> 1; j > 0; j >>= 1)
      bitonic_k<<<KP1 / 256, 256, 0, stream>>>(KEYS, j, k, KP1);
  mark_k<<<1, 256, 0, stream>>>(KEYS, MAP2, KEPT2, KP2);
  gather_k<<<nb((long long)KP2 * F3, 256), 256, 0, stream>>>(KEYS, AGG3, X4, (long long)KP2 * F3, 8);

  // ---- FC: out = fcb + flatten(x4^T) @ fcW ----
  init_out_k<<<1, 512, 0, stream>>>(fcb, out);
  fc_k<<<(KP2 * F3) / 512, 128, 0, stream>>>(X4, fcW, out);
}
